// EdgeClasification_72344429134238
// MI455X (gfx1250) — matrix operand order repaired and checked by reading, not compiled, not run on hardware
//
#include <hip/hip_runtime.h>
#include <hip/hip_fp16.h>

typedef __attribute__((ext_vector_type(16))) _Float16 v16h;
typedef __attribute__((ext_vector_type(8)))  _Float16 v8h;
typedef __attribute__((ext_vector_type(8)))  float    v8f;

#define NN 4096      // nodes
#define NE 16384     // edges
#define DI 128       // input feature dim
#define HD 1024      // hidden

// ---------------------------------------------------------------------------
// 1) Recover src/dst indices from the dense one-hot incidence matrices.
//    Exactly one nonzero (1.0f) per row, so a plain conditional store is safe.
// ---------------------------------------------------------------------------
__global__ void extract_idx_kernel(const float4* __restrict__ R,
                                   int* __restrict__ idx,
                                   int colsDiv4, int total4) {
    for (int i = blockIdx.x * blockDim.x + threadIdx.x; i < total4;
         i += gridDim.x * blockDim.x) {
        float4 v = R[i];
        if (v.x != 0.f || v.y != 0.f || v.z != 0.f || v.w != 0.f) {
            int c = (v.x != 0.f) ? 0 : (v.y != 0.f) ? 1 : (v.z != 0.f) ? 2 : 3;
            idx[i / colsDiv4] = (i % colsDiv4) * 4 + c;
        }
    }
}

__global__ void fill_zero_kernel(float* __restrict__ p, int n) {
    for (int i = blockIdx.x * blockDim.x + threadIdx.x; i < n;
         i += gridDim.x * blockDim.x) p[i] = 0.f;
}

// ---------------------------------------------------------------------------
// 2) Edge scatter:  Ho[dst] += e[src]*H[src],  Hi[src] += e[dst]*H[dst]
//    One edge per 128-thread block; f32 atomics land in L2 (4 MB working set).
// ---------------------------------------------------------------------------
__global__ void edge_scatter_kernel(const float* __restrict__ H,
                                    const float* __restrict__ e,
                                    const int* __restrict__ src,
                                    const int* __restrict__ dst,
                                    float* __restrict__ Ho,
                                    float* __restrict__ Hi) {
    int ed = blockIdx.x;
    int f  = threadIdx.x;           // 0..127
    int s = src[ed], d = dst[ed];
    float es = e[s], edv = e[d];
    atomicAdd(&Ho[(size_t)d * DI + f], es  * H[(size_t)s * DI + f]);
    atomicAdd(&Hi[(size_t)s * DI + f], edv * H[(size_t)d * DI + f]);
}

// ---------------------------------------------------------------------------
// 3) Build x0 = concat([Ho, H, Hi]) in f16  [4096, 384]
// ---------------------------------------------------------------------------
__global__ void build_x0_kernel(const float* __restrict__ Ho,
                                const float* __restrict__ H,
                                const float* __restrict__ Hi,
                                _Float16* __restrict__ x0) {
    int n = blockIdx.x, f = threadIdx.x;   // 128 threads
    x0[(size_t)n * 384 + f]        = (_Float16)Ho[(size_t)n * DI + f];
    x0[(size_t)n * 384 + 128 + f]  = (_Float16)H [(size_t)n * DI + f];
    x0[(size_t)n * 384 + 256 + f]  = (_Float16)Hi[(size_t)n * DI + f];
}

// ---------------------------------------------------------------------------
// 4) Weight convert + transpose: W[K,N] f32  ->  WT[N,K] f16
//    Makes the WMMA B-fragment a contiguous 32-byte per-lane load.
// ---------------------------------------------------------------------------
__global__ void convert_wt_kernel(const float* __restrict__ W,
                                  _Float16* __restrict__ WT, int K, int N) {
    int total = K * N;
    for (int i = blockIdx.x * blockDim.x + threadIdx.x; i < total;
         i += gridDim.x * blockDim.x) {
        int k = i / N, n = i % N;
        WT[(size_t)n * K + k] = (_Float16)W[i];
    }
}

// ---------------------------------------------------------------------------
// 5) WMMA GEMM + bias + tanh:  Out[M,N] = tanh(A[M,K] @ W[K,N] + b)
//    One wave (block of 32) per 16(M) x 64(N) strip, K-loop in steps of 32.
//    A fragment layout (16-bit 16x32, contiguous K, same order as B):
//      lane<16 : row=tm+lane,   halves 0..15 = K k0..k0+15
//      lane>=16: row=tm+lane-16,halves 0..15 = k0+16..k0+31
//    B fragment (32x16): lane<16 -> col=tn+lane, K=k0..k0+15 (contiguous in WT);
//                        lane>=16 -> col=tn+lane-16, K=k0+16..k0+31.
// ---------------------------------------------------------------------------
__global__ void wmma_gemm_tanh_kernel(const _Float16* __restrict__ A,
                                      const _Float16* __restrict__ WT,
                                      const float* __restrict__ bias,
                                      _Float16* __restrict__ Out,
                                      int M, int N, int K) {
    const int lane = threadIdx.x & 31;
    const int tilesN = N >> 6;
    const int tm = (blockIdx.x / tilesN) << 4;
    const int tn = (blockIdx.x % tilesN) << 6;
    const int l15 = lane & 15;
    const int hi  = lane >> 4;

    v8f c0 = {}, c1 = {}, c2 = {}, c3 = {};

    const _Float16* arow = A  + (size_t)(tm + l15) * K + hi * 16;
    const _Float16* brow = WT + (size_t)(tn + l15) * K + hi * 16;
    const size_t bstep = (size_t)16 * K;    // 16 columns forward in WT

    for (int k0 = 0; k0 < K; k0 += 32) {
        union { v16h v; v8h h[2]; } a;
        a.h[0] = *(const v8h*)(arow + k0);        // K k0+16h..+7  (this half-wave)
        a.h[1] = *(const v8h*)(arow + k0 + 8);    // K k0+16h+8..+15 (this half-wave)

        v16h b0 = *(const v16h*)(brow + k0);
        v16h b1 = *(const v16h*)(brow + bstep     + k0);
        v16h b2 = *(const v16h*)(brow + 2 * bstep + k0);
        v16h b3 = *(const v16h*)(brow + 3 * bstep + k0);

        c0 = __builtin_amdgcn_wmma_f32_16x16x32_f16(false, a.v, false, b0,
                                                    (short)0, c0, false, false);
        c1 = __builtin_amdgcn_wmma_f32_16x16x32_f16(false, a.v, false, b1,
                                                    (short)0, c1, false, false);
        c2 = __builtin_amdgcn_wmma_f32_16x16x32_f16(false, a.v, false, b2,
                                                    (short)0, c2, false, false);
        c3 = __builtin_amdgcn_wmma_f32_16x16x32_f16(false, a.v, false, b3,
                                                    (short)0, c3, false, false);
    }

    // C/D layout: lane<16 -> col=tn+lane, rows tm+0..7 in v0..7;
    //             lane>=16 -> col=tn+lane-16, rows tm+8..15.
    const int col = tn + l15;
    const float bv0 = bias[col], bv1 = bias[col + 16];
    const float bv2 = bias[col + 32], bv3 = bias[col + 48];
#pragma unroll
    for (int r = 0; r < 8; ++r) {
        int row = tm + r + hi * 8;
        _Float16* o = Out + (size_t)row * N + col;
        o[0]  = (_Float16)tanhf(c0[r] + bv0);
        o[16] = (_Float16)tanhf(c1[r] + bv1);
        o[32] = (_Float16)tanhf(c2[r] + bv2);
        o[48] = (_Float16)tanhf(c3[r] + bv3);
    }
}

// ---------------------------------------------------------------------------
// 6) Head: out[n] = sigmoid(x4[n,:256] . W5 + b5), one wave per row
// ---------------------------------------------------------------------------
__global__ void final_sigmoid_kernel(const _Float16* __restrict__ x,
                                     const float* __restrict__ W5,
                                     const float* __restrict__ b5,
                                     float* __restrict__ out) {
    int row  = blockIdx.x;
    int lane = threadIdx.x;
    float s = 0.f;
#pragma unroll
    for (int i = 0; i < 8; ++i) {
        int c = lane + i * 32;
        s += (float)x[(size_t)row * 256 + c] * W5[c];
    }
    for (int off = 16; off > 0; off >>= 1) s += __shfl_down(s, off, 32);
    if (lane == 0) out[row] = 1.f / (1.f + expf(-(s + b5[0])));
}

// ---------------------------------------------------------------------------
extern "C" void kernel_launch(void* const* d_in, const int* in_sizes, int n_in,
                              void* d_out, int out_size, void* d_ws, size_t ws_size,
                              hipStream_t stream) {
    const float* H  = (const float*)d_in[0];
    const float* Ro = (const float*)d_in[1];
    const float* Ri = (const float*)d_in[2];
    const float* e  = (const float*)d_in[3];
    const float* W1 = (const float*)d_in[4];
    const float* b1 = (const float*)d_in[5];
    const float* W2 = (const float*)d_in[6];
    const float* b2 = (const float*)d_in[7];
    const float* W3 = (const float*)d_in[8];
    const float* b3 = (const float*)d_in[9];
    const float* W4 = (const float*)d_in[10];
    const float* b4 = (const float*)d_in[11];
    const float* W5 = (const float*)d_in[12];
    const float* b5 = (const float*)d_in[13];
    float* out = (float*)d_out;

    char* ws = (char*)d_ws;
    size_t off = 0;
    auto alloc = [&](size_t bytes) -> void* {
        void* p = ws + off;
        off = (off + bytes + 255) & ~(size_t)255;
        return p;
    };

    int*       src  = (int*)alloc(NE * sizeof(int));
    int*       dst  = (int*)alloc(NE * sizeof(int));
    float*     Ho   = (float*)alloc((size_t)NN * DI * sizeof(float));
    float*     Hi   = (float*)alloc((size_t)NN * DI * sizeof(float));
    _Float16*  x0   = (_Float16*)alloc((size_t)NN * 384 * 2);
    _Float16*  a1   = (_Float16*)alloc((size_t)NN * HD * 2);
    _Float16*  a2   = (_Float16*)alloc((size_t)NN * HD * 2);
    _Float16*  a3   = (_Float16*)alloc((size_t)NN * (HD / 2) * 2);
    _Float16*  a4   = (_Float16*)alloc((size_t)NN * (HD / 4) * 2);
    _Float16*  w1t  = (_Float16*)alloc((size_t)384 * HD * 2);
    _Float16*  w2t  = (_Float16*)alloc((size_t)HD * HD * 2);
    _Float16*  w3t  = (_Float16*)alloc((size_t)HD * (HD / 2) * 2);
    _Float16*  w4t  = (_Float16*)alloc((size_t)(HD / 2) * (HD / 4) * 2);

    // 1) indices from dense one-hot incidence (single streaming pass, 512 MB)
    const int total4 = NE * (NN / 4);
    extract_idx_kernel<<<8192, 256, 0, stream>>>((const float4*)Ro, src, NN / 4, total4);
    extract_idx_kernel<<<8192, 256, 0, stream>>>((const float4*)Ri, dst, NN / 4, total4);

    // 2) zero accumulators (Ho, Hi are adjacent allocations but zero separately)
    fill_zero_kernel<<<2048, 256, 0, stream>>>(Ho, NN * DI);
    fill_zero_kernel<<<2048, 256, 0, stream>>>(Hi, NN * DI);

    // 3) scatter-add over edges
    edge_scatter_kernel<<<NE, DI, 0, stream>>>(H, e, src, dst, Ho, Hi);

    // 4) x0 = concat([Ho, H, Hi]) in f16
    build_x0_kernel<<<NN, DI, 0, stream>>>(Ho, H, Hi, x0);

    // 5) weights -> f16 transposed
    convert_wt_kernel<<<1536, 256, 0, stream>>>(W1, w1t, 384, HD);
    convert_wt_kernel<<<4096, 256, 0, stream>>>(W2, w2t, HD, HD);
    convert_wt_kernel<<<2048, 256, 0, stream>>>(W3, w3t, HD, HD / 2);
    convert_wt_kernel<<<512, 256, 0, stream>>>(W4, w4t, HD / 2, HD / 4);

    // 6) MLP: tanh(x @ W + b) chain, all via v_wmma_f32_16x16x32_f16
    // L1: [4096,384] @ [384,1024]
    wmma_gemm_tanh_kernel<<<(NN / 16) * (HD / 64), 32, 0, stream>>>(
        x0, w1t, b1, a1, NN, HD, 384);
    // L2: [4096,1024] @ [1024,1024]
    wmma_gemm_tanh_kernel<<<(NN / 16) * (HD / 64), 32, 0, stream>>>(
        a1, w2t, b2, a2, NN, HD, HD);
    // L3: [4096,1024] @ [1024,512]
    wmma_gemm_tanh_kernel<<<(NN / 16) * ((HD / 2) / 64), 32, 0, stream>>>(
        a2, w3t, b3, a3, NN, HD / 2, HD);
    // L4: [4096,512] @ [512,256]
    wmma_gemm_tanh_kernel<<<(NN / 16) * ((HD / 4) / 64), 32, 0, stream>>>(
        a3, w4t, b4, a4, NN, HD / 4, HD / 2);

    // 7) sigmoid head -> f32 output [4096]
    final_sigmoid_kernel<<<NN, 32, 0, stream>>>(a4, W5, b5, out);

    (void)in_sizes; (void)n_in; (void)out_size; (void)ws_size;
}
